// NeuralDDE_46823733461188
// MI455X (gfx1250) — matrix operand order repaired and checked by reading, not compiled, not run on hardware
//
#include <hip/hip_runtime.h>

typedef __attribute__((ext_vector_type(16))) _Float16 v16h;
typedef __attribute__((ext_vector_type(8)))  _Float16 v8h;
typedef __attribute__((ext_vector_type(2)))  __fp16   v2fp;
typedef __attribute__((ext_vector_type(8)))  float    v8f;

#define T_STEPS 256
#define BATCH   2048
#define DATA    32
#define WIDTH   128
#define IN_SZ   96          // (len(DELAYS)+1)*DATA
#define SWAPX16 0x401f      // ds_swizzle group-of-32: xor=0x10, or=0, and=0x1f

union BFrag { v16h v; unsigned u[8]; };
union AFr   { v16h v; v8h h[2]; };

// single-instruction f16 pair pack: v_cvt_pk_rtz_f16_f32
__device__ __forceinline__ unsigned pack2h(float a, float b) {
  union { v2fp h; unsigned u; } t;
  t.h = __builtin_amdgcn_cvt_pkrtz(a, b);
  return t.u;
}

// pack + fused ReLU on the packed pair: v_cvt_pk_rtz + v_pk_max_num_f16
__device__ __forceinline__ unsigned pack2h_relu(float a, float b) {
  union { v2fp h; unsigned u; } t;
  t.h = __builtin_amdgcn_cvt_pkrtz(a, b);
  v2fp z = {(__fp16)0.0f, (__fp16)0.0f};
  t.h = __builtin_elementwise_max(t.h, z);
  return t.u;
}

// ---- A operand = 16x(KT*32) tile of W (row-major [N][K], f16, global) ----
// A 16x32 (contiguous k): lanes 0-15 -> M=lane, V0..7=K 0..15
//                        lanes16-31 -> M=lane-16, V0..7=K 16..31
__device__ __forceinline__ v16h g_load_WA(const _Float16* __restrict__ W, int K,
                                          int mt, int kt) {
  const int lane = threadIdx.x & 31;
  const int m  = lane & 15;
  const int kb = ((lane >> 4) << 4) + kt * 32;
  const _Float16* p = W + (size_t)(mt * 16 + m) * K + kb;
  AFr f;
  f.h[0] = *(const v8h*)(p);
  f.h[1] = *(const v8h*)(p + 8);
  return f.v;
}

// ---- B fragment (one 32-K tile of Xt) from a PAIR of D-layout tiles ------
// D tile: lane -> batch col n=lane%16, VGPR r -> row m = r + 8*(lane/16).
// B frag: lane -> col n=lane%16, K = 16*(lane/16) + j, pairs packed per VGPR.
// Upper/lower row halves exchanged with ds_swizzle SWAPX16 (register-only).
template<bool RELU>
__device__ __forceinline__ v16h bfrag_from_pair(const v8f d0, const v8f d1) {
  const int lane = threadIdx.x & 31;
  const bool hi = lane >= 16;
  BFrag b;
  #pragma unroll
  for (int q = 0; q < 4; ++q) {
    unsigned p0 = RELU ? pack2h_relu(d0[2*q], d0[2*q+1]) : pack2h(d0[2*q], d0[2*q+1]);
    unsigned p1 = RELU ? pack2h_relu(d1[2*q], d1[2*q+1]) : pack2h(d1[2*q], d1[2*q+1]);
    unsigned s0 = (unsigned)__builtin_amdgcn_ds_swizzle((int)p0, SWAPX16);
    unsigned s1 = (unsigned)__builtin_amdgcn_ds_swizzle((int)p1, SWAPX16);
    b.u[q]     = hi ? s1 : p0;   // j = 2q, 2q+1
    b.u[4 + q] = hi ? p1 : s0;   // j = 8+2q, 9+2q
  }
  return b.v;
}

// ---- history B fragment: lane = batch row, K = data idx (vector loads) ----
__device__ __forceinline__ v16h load_hist_B(const float* __restrict__ ys, int row0,
                                            float tt, float tau, float t0, float rdt0) {
  float s = (tt - tau - t0) * rdt0;                 // rdt0 hoisted: no div here
  s = fminf(fmaxf(s, 0.0f), (float)(T_STEPS - 1));
  int i0 = (int)s;                                  // s >= 0 -> trunc == floor
  int i1 = i0 + 1; if (i1 > T_STEPS - 1) i1 = T_STEPS - 1;
  float w = s - (float)i0;
  const int lane = threadIdx.x & 31;
  const int brow = row0 + (lane & 15);
  const int koff = (lane >> 4) << 4;
  const float* p0 = ys + ((size_t)i0 * BATCH + brow) * DATA + koff;
  const float* p1 = ys + ((size_t)i1 * BATCH + brow) * DATA + koff;
  BFrag b;
  #pragma unroll
  for (int q = 0; q < 4; ++q) {
    float4 a0 = *(const float4*)(p0 + 4 * q);
    float4 a1 = *(const float4*)(p1 + 4 * q);
    float v0 = (1.0f - w) * a0.x + w * a1.x;
    float v1 = (1.0f - w) * a0.y + w * a1.y;
    float v2 = (1.0f - w) * a0.z + w * a1.z;
    float v3 = (1.0f - w) * a0.w + w * a1.w;
    b.u[2*q]   = pack2h(v0, v1);
    b.u[2*q+1] = pack2h(v2, v3);
  }
  return b.v;
}

// ---- fused hidden layer: produce M-tile PAIRS, immediately fold into the
//      next layer's B fragments (only 2 D tiles ever live); ReLU is fused
//      into the f16 pack (v_pk_max_num_f16) ------------------------------
template<int KT>
__device__ __forceinline__ void layer_fused(const v16h* bin,
                                            const _Float16* W,
                                            const float* bias,
                                            v16h bout[4]) {
  const int lane = threadIdx.x & 31;
  const int mb8 = (lane >> 4) << 3;
  #pragma unroll
  for (int p = 0; p < 4; ++p) {                   // pair of 16-row W tiles
    v8f c0, c1;
    #pragma unroll
    for (int half = 0; half < 2; ++half) {
      const int mt = 2 * p + half;
      const float* bp = bias + mt * 16 + mb8;     // bias along M: contiguous
      float4 blo = *(const float4*)(bp);
      float4 bhi = *(const float4*)(bp + 4);
      v8f c = {blo.x, blo.y, blo.z, blo.w, bhi.x, bhi.y, bhi.z, bhi.w};
      #pragma unroll
      for (int kt = 0; kt < KT; ++kt) {
        v16h a = g_load_WA(W, KT * 32, mt, kt);
        c = __builtin_amdgcn_wmma_f32_16x16x32_f16(false, a, false, bin[kt],
                                                   (short)0, c, false, false);
      }
      if (half == 0) c0 = c; else c1 = c;
    }
    bout[p] = bfrag_from_pair<true>(c0, c1);      // relu fused in the pack
  }
}

// ---- final layer: 2 output tiles, keep D layout (no relu) ----------------
__device__ __forceinline__ void layer_out(const v16h* bin,
                                          const _Float16* W,
                                          const float* bias,
                                          v8f k[2]) {
  const int lane = threadIdx.x & 31;
  const int mb8 = (lane >> 4) << 3;
  #pragma unroll
  for (int mt = 0; mt < 2; ++mt) {
    const float* bp = bias + mt * 16 + mb8;
    float4 blo = *(const float4*)(bp);
    float4 bhi = *(const float4*)(bp + 4);
    v8f c = {blo.x, blo.y, blo.z, blo.w, bhi.x, bhi.y, bhi.z, bhi.w};
    #pragma unroll
    for (int kt = 0; kt < 4; ++kt) {
      v16h a = g_load_WA(W, 128, mt, kt);
      c = __builtin_amdgcn_wmma_f32_16x16x32_f16(false, a, false, bin[kt],
                                                 (short)0, c, false, false);
    }
    k[mt] = c;
  }
}

// ---- f(t, y + adt*kp): full MLP, register-resident -----------------------
__device__ __forceinline__ void eval_f(const v8f y[2], const v8f kp[2], float adt,
    float tt, const float* __restrict__ ys, int row0, float t0, float rdt0,
    const _Float16* W0h, const _Float16* W1h,
    const _Float16* W2h, const _Float16* W3h,
    const float* b0, const float* b1, const float* b2, const float* b3,
    v8f kout[2]) {
  // K-tile 0: ye = y + adt*kp, folded directly into the pack (no ye array)
  v8f e0, e1;
  #pragma unroll
  for (int r = 0; r < 8; ++r) { e0[r] = y[0][r] + adt * kp[0][r];
                                e1[r] = y[1][r] + adt * kp[1][r]; }
  v16h bx[3];
  bx[0] = bfrag_from_pair<false>(e0, e1);
  bx[1] = load_hist_B(ys, row0, tt, 1.0f, t0, rdt0);   // hist(t-1)
  bx[2] = load_hist_B(ys, row0, tt, 2.0f, t0, rdt0);   // hist(t-2)

  v16h bA[4], bB[4];
  layer_fused<3>(bx, W0h, b0, bA);
  layer_fused<4>(bA, W1h, b1, bB);
  layer_fused<4>(bB, W2h, b2, bA);
  layer_out(bA, W3h, b3, kout);
}

// ---- persistent solver: one wave owns 16 batch rows for all 255 steps ----
__global__ __launch_bounds__(32)
__attribute__((amdgpu_waves_per_eu(4)))
void dde_solve_kernel(
    const float* __restrict__ ts, const float* __restrict__ y0,
    const float* __restrict__ b0, const float* __restrict__ b1,
    const float* __restrict__ b2, const float* __restrict__ b3,
    const _Float16* __restrict__ Wh, float* __restrict__ ys) {
  const _Float16* W0h = Wh;
  const _Float16* W1h = W0h + WIDTH * IN_SZ;
  const _Float16* W2h = W1h + WIDTH * WIDTH;
  const _Float16* W3h = W2h + WIDTH * WIDTH;

  const int lane = threadIdx.x & 31;
  const int row0 = blockIdx.x * 16;
  const int n   = lane & 15;
  const int mb8 = (lane >> 4) << 3;

  const float t0   = ts[0];
  const float dt0  = ts[1] - ts[0];
  const float rdt0 = 1.0f / dt0;                 // one divide, outside loop

  // y state in D layout: lane = batch col, VGPR = data idx (vector loads)
  v8f y[2];
  {
    const float* src = y0 + (size_t)(row0 + n) * DATA + mb8;
    #pragma unroll
    for (int mt = 0; mt < 2; ++mt) {
      float4 lo = *(const float4*)(src + mt * 16);
      float4 hi = *(const float4*)(src + mt * 16 + 4);
      y[mt] = {lo.x, lo.y, lo.z, lo.w, hi.x, hi.y, hi.z, hi.w};
    }
  }

  for (int i = 0; i < T_STEPS - 1; ++i) {
    const float t  = ts[i];
    const float dt = ts[i + 1] - t;
    v8f acc[2], k[2];
    acc[0] = y[0]; acc[1] = y[1];
    #pragma unroll
    for (int nt = 0; nt < 2; ++nt)
      #pragma unroll
      for (int r = 0; r < 8; ++r) k[nt][r] = 0.0f;
    float adt = 0.0f;                            // stage-0 input is y itself

    #pragma unroll 1          // ONE copy of the 96-WMMA stage body
    for (int s = 0; s < 3; ++s) {
      const float cs = (s == 0) ? 0.0f : (s == 1 ? 0.5f : 0.75f);
      const float wsn = (s == 0) ? (2.0f/9.0f) : (s == 1 ? (1.0f/3.0f) : (4.0f/9.0f));

      // Launder the loop-invariant weight/bias pointers so LICM cannot hoist
      // their loads out of the stage loop (which spilled them to scratch).
      // Re-issued loads hit L0/L2; scratch traffic disappears.
      const _Float16 *w0 = W0h, *w1 = W1h, *w2 = W2h, *w3 = W3h;
      const float *p0 = b0, *p1 = b1, *p2 = b2, *p3 = b3;
      asm volatile("" : "+s"(w0), "+s"(w1), "+s"(w2), "+s"(w3),
                        "+s"(p0), "+s"(p1), "+s"(p2), "+s"(p3));

      v8f knew[2];
      eval_f(y, k, adt, t + cs * dt, ys, row0, t0, rdt0,
             w0, w1, w2, w3, p0, p1, p2, p3, knew);
      const float wdt = dt * wsn;
      #pragma unroll
      for (int nt = 0; nt < 2; ++nt)
        #pragma unroll
        for (int r = 0; r < 8; ++r) {
          k[nt][r] = knew[nt][r];
          acc[nt][r] += wdt * knew[nt][r];
        }
      adt = dt * ((s == 0) ? 0.5f : 0.75f);      // for next stage's ye
    }

    y[0] = acc[0]; y[1] = acc[1];
    // store y_{i+1}: per-lane contiguous along data dim -> float4 stores
    float* dst = ys + ((size_t)(i + 1) * BATCH + row0 + n) * DATA + mb8;
    #pragma unroll
    for (int mt = 0; mt < 2; ++mt) {
      *(float4*)(dst + mt * 16)     = make_float4(y[mt][0], y[mt][1], y[mt][2], y[mt][3]);
      *(float4*)(dst + mt * 16 + 4) = make_float4(y[mt][4], y[mt][5], y[mt][6], y[mt][7]);
    }
    // history writes drained before far-future reads (same-wave anyway)
    asm volatile("s_wait_storecnt 0" ::: "memory");
  }
}

// ---- one-time prep: f32 weights -> f16 in d_ws -------------------------
__global__ void cvt_weights_kernel(const float* __restrict__ W0, const float* __restrict__ W1,
                                   const float* __restrict__ W2, const float* __restrict__ W3,
                                   _Float16* __restrict__ out) {
  const int i = blockIdx.x * blockDim.x + threadIdx.x;
  const int n0 = WIDTH * IN_SZ, n1 = WIDTH * WIDTH, n2 = WIDTH * WIDTH, n3 = DATA * WIDTH;
  if (i < n0)                      out[i] = (_Float16)W0[i];
  else if (i < n0 + n1)            out[i] = (_Float16)W1[i - n0];
  else if (i < n0 + n1 + n2)       out[i] = (_Float16)W2[i - n0 - n1];
  else if (i < n0 + n1 + n2 + n3)  out[i] = (_Float16)W3[i - n0 - n1 - n2];
}

// ---- init: ys[0] = y0, trailing int32 output = T-1 ---------------------
__global__ void init_out_kernel(const float* __restrict__ y0, float* __restrict__ out,
                                int out_size) {
  const int i = blockIdx.x * blockDim.x + threadIdx.x;
  if (i < BATCH * DATA) out[i] = y0[i];
  if (i == 0) ((int*)out)[out_size - 1] = T_STEPS - 1;
}

extern "C" void kernel_launch(void* const* d_in, const int* in_sizes, int n_in,
                              void* d_out, int out_size, void* d_ws, size_t ws_size,
                              hipStream_t stream) {
  const float* ts = (const float*)d_in[0];
  const float* y0 = (const float*)d_in[1];
  const float* W0 = (const float*)d_in[2];
  const float* b0 = (const float*)d_in[3];
  const float* W1 = (const float*)d_in[4];
  const float* b1 = (const float*)d_in[5];
  const float* W2 = (const float*)d_in[6];
  const float* b2 = (const float*)d_in[7];
  const float* W3 = (const float*)d_in[8];
  const float* b3 = (const float*)d_in[9];
  _Float16* Wh = (_Float16*)d_ws;
  float* out = (float*)d_out;

  const int wtot = WIDTH * IN_SZ + 2 * WIDTH * WIDTH + DATA * WIDTH;
  cvt_weights_kernel<<<(wtot + 255) / 256, 256, 0, stream>>>(W0, W1, W2, W3, Wh);
  init_out_kernel<<<(BATCH * DATA + 255) / 256, 256, 0, stream>>>(y0, out, out_size);
  dde_solve_kernel<<<BATCH / 16, 32, 0, stream>>>(ts, y0, b0, b1, b2, b3, Wh, out);
}
